// SimpleGCNEncoder_12584254178049
// MI455X (gfx1250) — compile-verified, matrix operand order checked by reading, not run on hardware
//
#include <hip/hip_runtime.h>

// ---------------------------------------------------------------------------
// Types for CDNA5 WMMA (gfx1250, wave32)
// ---------------------------------------------------------------------------
typedef __bf16 bf16;
typedef bf16  v8bf  __attribute__((ext_vector_type(8)));
typedef bf16  v16bf __attribute__((ext_vector_type(16)));
typedef float v8f   __attribute__((ext_vector_type(8)));

#define N_NODES 100000
#define N_EDGES 3200000

// ---------------------------------------------------------------------------
// Utility kernels
// ---------------------------------------------------------------------------
__global__ void zero_f32_kernel(float* __restrict__ p, long long n) {
    long long i = (long long)blockIdx.x * blockDim.x + threadIdx.x;
    if (i < n) p[i] = 0.0f;
}

__global__ void f32_to_bf16_kernel(const float* __restrict__ in,
                                   bf16* __restrict__ out, long long n) {
    long long i = (long long)blockIdx.x * blockDim.x + threadIdx.x;
    if (i < n) out[i] = (bf16)in[i];
}

// Pack W (K x N, f32 row-major) into the per-lane WMMA-B bf16 layout:
// flat index = ((kt*ntiles + nt)*32 + lane)*16 + e
//   k = kt*32 + (lane>=16 ? 8 : 0) + (e>=8 ? 16 : 0) + (e&7)
//   n = nt*16 + (lane&15)
__global__ void pack_w_kernel(const float* __restrict__ W, bf16* __restrict__ Wp,
                              int K, int N) {
    int idx = blockIdx.x * blockDim.x + threadIdx.x;
    int ntiles = N >> 4;
    int total = (K >> 5) * ntiles * 512;
    if (idx >= total) return;
    int e    = idx & 15;
    int lane = (idx >> 4) & 31;
    int tile = idx >> 9;
    int nt = tile % ntiles;
    int kt = tile / ntiles;
    int k = kt * 32 + ((lane >> 4) << 3) + ((e >> 3) << 4) + (e & 7);
    int n = nt * 16 + (lane & 15);
    Wp[idx] = (bf16)W[k * N + n];
}

// ---------------------------------------------------------------------------
// GEMM: Z = A(bf16, MxK row-major) @ W(packed bf16) + bias,  Z f32 MxN.
// Block = 256 threads = 8 waves. Each wave computes a 32x64 output macro-tile:
// 2 M-tiles (pair index blockIdx.x*8+wave, M multiple of 32) x 4 N-tiles
// (nt0 = blockIdx.y*4). K stepped by 32 with V_WMMA_F32_16X16X32_BF16,
// f32 accumulation. Per K-step: 12 b128 loads feed 8 WMMAs (1.5 loads/WMMA);
// each B fragment is register-reused by two WMMAs.
// ---------------------------------------------------------------------------
__global__ __launch_bounds__(256)
void gemm_wmma_bf16_kernel(const bf16* __restrict__ A, const bf16* __restrict__ Wp,
                           const float* __restrict__ bias, float* __restrict__ Z,
                           int M, int N, int K) {
    const int lane = threadIdx.x & 31;
    const int wave = threadIdx.x >> 5;
    const int pair = blockIdx.x * 8 + wave;     // pair of 16-row M-tiles
    if (pair * 32 >= M) return;                  // wave-uniform (EXEC stays all-1)

    const int ntiles  = N >> 4;
    const int ktiles  = K >> 5;
    const int nt0     = blockIdx.y * 4;
    const int halfsel = lane >> 4;              // 0: K 0-7/16-23, 1: K 8-15/24-31
    const int mrow0   = pair * 32 + (lane & 15);
    const bf16* __restrict__ arow0 = A + (long long)mrow0 * K + (halfsel << 3);
    const bf16* __restrict__ arow1 = arow0 + (long long)16 * K;

    v8f acc[2][4];
#pragma unroll
    for (int t = 0; t < 2; ++t)
#pragma unroll
        for (int j = 0; j < 4; ++j) acc[t][j] = (v8f){};

    for (int kt = 0; kt < ktiles; ++kt) {
        // Two A fragments (ISA 16-bit A layout: two 16B groups per lane)
        v8bf a0l = *(const v8bf*)(arow0 + kt * 32);
        v8bf a0h = *(const v8bf*)(arow0 + kt * 32 + 16);
        v8bf a1l = *(const v8bf*)(arow1 + kt * 32);
        v8bf a1h = *(const v8bf*)(arow1 + kt * 32 + 16);
        v16bf a0 = __builtin_shufflevector(a0l, a0h,
            0, 1, 2, 3, 4, 5, 6, 7, 8, 9, 10, 11, 12, 13, 14, 15);
        v16bf a1 = __builtin_shufflevector(a1l, a1h,
            0, 1, 2, 3, 4, 5, 6, 7, 8, 9, 10, 11, 12, 13, 14, 15);

        const bf16* __restrict__ bbase =
            Wp + (((long long)(kt * ntiles + nt0) * 32) + lane) * 16;
#pragma unroll
        for (int j = 0; j < 4; ++j) {
            v16bf b = *(const v16bf*)(bbase + (long long)j * 32 * 16);
            acc[0][j] = __builtin_amdgcn_wmma_f32_16x16x32_bf16(
                false, a0, false, b, (short)0, acc[0][j], false, false);
            acc[1][j] = __builtin_amdgcn_wmma_f32_16x16x32_bf16(
                false, a1, false, b, (short)0, acc[1][j], false, false);
        }
    }

    // Epilogue: C/D layout -> element (M = halfsel*8 + r, N = lane&15) in VGPR r
    const int ncol = lane & 15;
#pragma unroll
    for (int t = 0; t < 2; ++t) {
        const int mbase = pair * 32 + t * 16 + (halfsel << 3);
#pragma unroll
        for (int j = 0; j < 4; ++j) {
            int n = (nt0 + j) * 16 + ncol;
            float bv = bias[n];
            float* __restrict__ zcol = Z + (long long)mbase * N + n;
#pragma unroll
            for (int r = 0; r < 8; ++r)
                zcol[(long long)r * N] = acc[t][j][r] + bv;
        }
    }
}

// ---------------------------------------------------------------------------
// SpMM (COO, scatter with hardware f32 atomics resolved in L2):
// out[row[e], :] += vals[e] * H[col[e], :]
// One wave per edge; lane owns DIM/32 contiguous floats (32B per lane).
// ---------------------------------------------------------------------------
template <int DIM>
__global__ __launch_bounds__(256)
void spmm_atomic_kernel(const int* __restrict__ rowi, const int* __restrict__ coli,
                        const float* __restrict__ vals, const float* __restrict__ H,
                        float* __restrict__ out, int n_edges) {
    const int lane = threadIdx.x & 31;
    const int e = blockIdx.x * (blockDim.x >> 5) + (threadIdx.x >> 5);
    if (e >= n_edges) return;
    const float v = vals[e];
    const int r = rowi[e];
    const int c = coli[e];
    constexpr int PER = DIM / 32;  // 8 (256-dim) or 4 (128-dim)
    const float* __restrict__ src = H + (long long)c * DIM + lane * PER;
    float* __restrict__ dst = out + (long long)r * DIM + lane * PER;
#pragma unroll
    for (int i = 0; i < PER; i += 4) {
        float4 h4 = *(const float4*)(src + i);
        unsafeAtomicAdd(dst + i + 0, v * h4.x);
        unsafeAtomicAdd(dst + i + 1, v * h4.y);
        unsafeAtomicAdd(dst + i + 2, v * h4.z);
        unsafeAtomicAdd(dst + i + 3, v * h4.w);
    }
}

// ---------------------------------------------------------------------------
// BatchNorm: per-column sum / sumsq reduction (C must equal blockDim.x = 256)
// ---------------------------------------------------------------------------
__global__ __launch_bounds__(256)
void bn_stats_kernel(const float* __restrict__ H, float* __restrict__ sums,
                     int M, int C) {
    const int c = threadIdx.x;
    const int r0 = blockIdx.x * 256;
    const int r1 = (r0 + 256 < M) ? (r0 + 256) : M;
    float s = 0.0f, s2 = 0.0f;
    for (int r = r0; r < r1; ++r) {
        float x = H[(long long)r * C + c];
        s += x;
        s2 += x * x;
    }
    unsafeAtomicAdd(&sums[c], s);
    unsafeAtomicAdd(&sums[C + c], s2);
}

__global__ void bn_finalize_kernel(const float* __restrict__ sums,
                                   const float* __restrict__ gamma,
                                   const float* __restrict__ beta,
                                   float* __restrict__ ss, int M, int C) {
    int c = blockIdx.x * blockDim.x + threadIdx.x;
    if (c >= C) return;
    float inv_m = 1.0f / (float)M;
    float mean = sums[c] * inv_m;
    float var = sums[C + c] * inv_m - mean * mean;
    float sc = gamma[c] * rsqrtf(var + 1e-5f);
    ss[c] = sc;
    ss[C + c] = beta[c] - mean * sc;
}

// Fused BN apply + ReLU + convert to bf16 (produces next GEMM's A operand)
__global__ void bn_relu_bf16_kernel(const float* __restrict__ H,
                                    const float* __restrict__ ss,
                                    bf16* __restrict__ out, long long total, int C) {
    long long i = (long long)blockIdx.x * blockDim.x + threadIdx.x;
    if (i >= total) return;
    int c = (int)(i & (C - 1));  // C = 256 (power of two)
    float y = H[i] * ss[c] + ss[C + c];
    out[i] = (bf16)fmaxf(y, 0.0f);
}

// ---------------------------------------------------------------------------
// Host orchestration
// ---------------------------------------------------------------------------
static inline size_t align_up(size_t x, size_t a) { return (x + a - 1) & ~(a - 1); }

extern "C" void kernel_launch(void* const* d_in, const int* in_sizes, int n_in,
                              void* d_out, int out_size, void* d_ws, size_t ws_size,
                              hipStream_t stream) {
    // Inputs in setup_inputs() order:
    const float* x    = (const float*)d_in[0];
    const float* vals = (const float*)d_in[1];
    const float* W0   = (const float*)d_in[2];
    const float* b0   = (const float*)d_in[3];
    const float* g0   = (const float*)d_in[4];
    const float* be0  = (const float*)d_in[5];
    const float* W1   = (const float*)d_in[6];
    const float* b1   = (const float*)d_in[7];
    const float* g1   = (const float*)d_in[8];
    const float* be1  = (const float*)d_in[9];
    const float* W2   = (const float*)d_in[10];
    const float* b2   = (const float*)d_in[11];
    const int*   row  = (const int*)d_in[12];
    const int*   col  = (const int*)d_in[13];
    float* out = (float*)d_out;

    const int M = N_NODES, E = N_EDGES;
    const long long MC = (long long)M * 256;

    // Workspace carve-out (~256 MB)
    char* ws = (char*)d_ws;
    size_t off = 0;
    float* Z = (float*)(ws + off);  off = align_up(off + MC * sizeof(float), 256);
    float* H = (float*)(ws + off);  off = align_up(off + MC * sizeof(float), 256);
    bf16*  Abf = (bf16*)(ws + off); off = align_up(off + MC * sizeof(bf16), 256);
    bf16*  Wp  = (bf16*)(ws + off); off = align_up(off + 256 * 256 * sizeof(bf16), 256);
    float* stats = (float*)(ws + off); off = align_up(off + 512 * sizeof(float), 256);
    float* ss    = (float*)(ws + off); off = align_up(off + 512 * sizeof(float), 256);
    (void)ws_size;

    const dim3 blk256(256);
    const int pair_blocks  = (M / 32 + 7) / 8;                  // 391 (3125 pairs)
    const int eblocks      = (E + 7) / 8;                       // 400000
    const int elem_blocks  = (int)((MC + 255) / 256);           // 100000
    const int stat_blocks  = (M + 255) / 256;                   // 391

    // ---- layer 0 ----
    f32_to_bf16_kernel<<<elem_blocks, blk256, 0, stream>>>(x, Abf, MC);
    pack_w_kernel<<<(256 / 32 * 256 / 16 * 512 + 255) / 256, blk256, 0, stream>>>(W0, Wp, 256, 256);
    gemm_wmma_bf16_kernel<<<dim3(pair_blocks, 4), blk256, 0, stream>>>(Abf, Wp, b0, Z, M, 256, 256);
    zero_f32_kernel<<<elem_blocks, blk256, 0, stream>>>(H, MC);
    spmm_atomic_kernel<256><<<eblocks, blk256, 0, stream>>>(row, col, vals, Z, H, E);
    zero_f32_kernel<<<2, blk256, 0, stream>>>(stats, 512);
    bn_stats_kernel<<<stat_blocks, blk256, 0, stream>>>(H, stats, M, 256);
    bn_finalize_kernel<<<1, blk256, 0, stream>>>(stats, g0, be0, ss, M, 256);
    bn_relu_bf16_kernel<<<elem_blocks, blk256, 0, stream>>>(H, ss, Abf, MC, 256);

    // ---- layer 1 ----
    pack_w_kernel<<<(256 / 32 * 256 / 16 * 512 + 255) / 256, blk256, 0, stream>>>(W1, Wp, 256, 256);
    gemm_wmma_bf16_kernel<<<dim3(pair_blocks, 4), blk256, 0, stream>>>(Abf, Wp, b1, Z, M, 256, 256);
    zero_f32_kernel<<<elem_blocks, blk256, 0, stream>>>(H, MC);
    spmm_atomic_kernel<256><<<eblocks, blk256, 0, stream>>>(row, col, vals, Z, H, E);
    zero_f32_kernel<<<2, blk256, 0, stream>>>(stats, 512);
    bn_stats_kernel<<<stat_blocks, blk256, 0, stream>>>(H, stats, M, 256);
    bn_finalize_kernel<<<1, blk256, 0, stream>>>(stats, g1, be1, ss, M, 256);
    bn_relu_bf16_kernel<<<elem_blocks, blk256, 0, stream>>>(H, ss, Abf, MC, 256);

    // ---- final projection (N = 128) ----
    pack_w_kernel<<<(256 / 32 * 128 / 16 * 512 + 255) / 256, blk256, 0, stream>>>(W2, Wp, 256, 128);
    gemm_wmma_bf16_kernel<<<dim3(pair_blocks, 2), blk256, 0, stream>>>(Abf, Wp, b2, Z, M, 128, 256);
    const long long out_elems = (long long)M * 128;
    zero_f32_kernel<<<(int)((out_elems + 255) / 256), blk256, 0, stream>>>(out, out_elems);
    spmm_atomic_kernel<128><<<eblocks, blk256, 0, stream>>>(row, col, vals, Z, out, E);
}
